// CWICLinear_40295383171455
// MI455X (gfx1250) — matrix operand order repaired and checked by reading, not compiled, not run on hardware
//
#include <hip/hip_runtime.h>

// ---------------------------------------------------------------------------
// CWIC stripe-masked linear, CDNA5 (gfx1250) wave32.
//  - Weight pre-transposed + converted once to bf16 [O][I] in scratch (if it
//    fits); hot loop streams B tiles with the Tensor Data Mover
//    (tensor_load_to_lds, double-buffered, s_wait_tensorcnt) straight into
//    LDS with hardware padding for conflict-free ds_load_b128 fragment reads.
//  - Fallback (scratch too small): TDM streams the raw f32 weight tile and
//    fragments are converted with hardware bf16 casts on read.
//  - Compute: v_wmma_f32_16x16x32_bf16, f32 accumulation over K=2048.
// ---------------------------------------------------------------------------

#define IN_F     2048
#define OUT_F    8192
#define STRIPE_W 256
#define NSTRIPES 32
#define NTOK     2048   // B*S
#define ROWS     32     // token rows per workgroup
#define KTILE    32     // K per WMMA step
#define ASTRIDE  40     // bf16 per A row in LDS (32 + pad, 16B-aligned rows)
#define BSTR     40     // bf16 per B row in LDS (32 + 4-DWORD TDM pad)

typedef __bf16          bf16_t;
typedef bf16_t          v16bf  __attribute__((ext_vector_type(16)));
typedef bf16_t          v8bf   __attribute__((ext_vector_type(8)));
typedef bf16_t          v4bf   __attribute__((ext_vector_type(4)));
typedef float           v8f    __attribute__((ext_vector_type(8)));
typedef unsigned int    v4u    __attribute__((ext_vector_type(4)));
typedef unsigned int    v8u    __attribute__((ext_vector_type(8)));

// Low 32 bits of a generic pointer to LDS == workgroup-relative LDS byte
// address (shared aperture lives entirely in addr[63:32]).
__device__ __forceinline__ unsigned lds_lo32(const void* p) {
    return (unsigned)(unsigned long long)(size_t)p;
}

// --------------------------- prep 1: pmb = mu@W + bias; zero counters ------
__global__ __launch_bounds__(256) void cwic_prep(
    const float* __restrict__ weight, const float* __restrict__ bias,
    const float* __restrict__ mu, float* __restrict__ pmb,
    int* __restrict__ counts)
{
    const int o = blockIdx.x * 256 + threadIdx.x;
    if (o < NTOK) counts[o] = 0;
    if (o >= OUT_F) return;
    float a0 = 0.f, a1 = 0.f, a2 = 0.f, a3 = 0.f;
    for (int i = 0; i < IN_F; i += 4) {
        a0 += mu[i + 0] * weight[(size_t)(i + 0) * OUT_F + o];
        a1 += mu[i + 1] * weight[(size_t)(i + 1) * OUT_F + o];
        a2 += mu[i + 2] * weight[(size_t)(i + 2) * OUT_F + o];
        a3 += mu[i + 3] * weight[(size_t)(i + 3) * OUT_F + o];
    }
    pmb[o] = bias[o] + ((a0 + a1) + (a2 + a3));
}

// --------------------------- prep 2: W[I][O] f32 -> Wt[O][I] bf16 ----------
__global__ __launch_bounds__(256) void cwic_wt_bf16(
    const float* __restrict__ weight, bf16_t* __restrict__ wt)
{
    __shared__ float tile[32][65];
    const int o0 = blockIdx.x * 64;
    const int i0 = blockIdx.y * 32;
    const int t  = threadIdx.x;
#pragma unroll
    for (int p = 0; p < 8; ++p) {
        const int idx = p * 256 + t;
        const int il = idx >> 6, ol = idx & 63;
        tile[il][ol] = weight[(size_t)(i0 + il) * OUT_F + o0 + ol];
    }
    __syncthreads();
#pragma unroll
    for (int p = 0; p < 8; ++p) {
        const int idx = p * 256 + t;
        const int ol = idx >> 5, il = idx & 31;
        wt[(size_t)(o0 + ol) * IN_F + i0 + il] = (bf16_t)tile[il][ol];
    }
}

// --------------------------- main stripe GEMM ------------------------------
template <bool PRE>
__global__ __launch_bounds__(256) void cwic_stripe_gemm(
    const float* __restrict__ x,
    const float* __restrict__ weight,          // f32 [I][O]   (PRE=false)
    const bf16_t* __restrict__ wtp,            // bf16 [O][I]  (PRE=true)
    const float* __restrict__ mu, const float* __restrict__ stdv,
    const float* __restrict__ thresholds, const float* __restrict__ pmb,
    int* __restrict__ counts, float* __restrict__ y)
{
    constexpr unsigned BUFBYTES =
        PRE ? (STRIPE_W * BSTR * 2u)          // 256 rows * 80 B (TDM-padded)
            : (KTILE * STRIPE_W * 4u);        // 32 rows  * 1 KB f32

    __shared__ __align__(16) float         s_thr[IN_F];
    __shared__ __align__(16) float         s_mu [IN_F];
    __shared__ __align__(16) bf16_t        s_A  [ROWS * ASTRIDE];
    __shared__ __align__(16) unsigned char s_Braw[2 * BUFBYTES];
    __shared__ int                         s_cnt[ROWS];

    const int r     = blockIdx.x;
    const int nbase = blockIdx.y * ROWS;
    const int t     = threadIdx.x;

    const unsigned ldsB0 = lds_lo32(&s_Braw[0]);

    // Issue one TDM load of the (next) B tile into LDS buffer `buf`.
    auto tdm_issue = [&](int k0n, int buf) {
        unsigned long long ga;
        v8u g1;
        if constexpr (PRE) {
            ga = (unsigned long long)(size_t)wtp +
                 2ull * ((unsigned long long)r * STRIPE_W * IN_F + (unsigned)k0n);
            // 2B elements, pad_enable, pad_interval=16 DW, pad_amount=4 DW
            g1[0] = (1u << 16) | (1u << 20) | (3u << 22) | (3u << 25);
            g1[1] = ((unsigned)IN_F & 0xffffu) << 16;        // tensor_dim0 lo
            g1[2] = ((unsigned)STRIPE_W & 0xffffu) << 16;    // dim0 hi | tensor_dim1 lo
            g1[3] = ((unsigned)KTILE) << 16;                 // dim1 hi | tile_dim0 = 32
            g1[4] = (unsigned)STRIPE_W;                      // tile_dim1 = 256 (tile_dim2 = 0)
            g1[5] = (unsigned)IN_F;                          // tensor_dim0_stride lo32
            g1[6] = 0u; g1[7] = 0u;
        } else {
            ga = (unsigned long long)(size_t)weight +
                 4ull * ((unsigned long long)k0n * OUT_F + (unsigned)(r * STRIPE_W));
            g1[0] = (2u << 16);                              // 4B elements, no pad
            g1[1] = ((unsigned)OUT_F & 0xffffu) << 16;       // tensor_dim0 lo
            g1[2] = ((unsigned)OUT_F >> 16) |
                    (((unsigned)KTILE & 0xffffu) << 16);     // dim0 hi | tensor_dim1 lo
            g1[3] = ((unsigned)STRIPE_W) << 16;              // tile_dim0 = 256
            g1[4] = (unsigned)KTILE;                         // tile_dim1 = 32
            g1[5] = (unsigned)OUT_F;                         // tensor_dim0_stride lo32
            g1[6] = 0u; g1[7] = 0u;
        }
        v4u g0;
        g0[0] = 1u;                                          // count=1, user load
        g0[1] = ldsB0 + (unsigned)buf * BUFBYTES;            // lds_addr
        g0[2] = (unsigned)ga;                                // global_addr lo32
        g0[3] = (unsigned)((ga >> 32) & 0x01ffffffull) | (2u << 30); // hi | type=2
        v4u gz = {0u, 0u, 0u, 0u};
        asm volatile("tensor_load_to_lds %0, %1, %2, %3"
                     :: "s"(g0), "s"(g1), "s"(gz), "s"(gz)
                     : "memory");
    };

    const int wave = t >> 5;
    const int lane = t & 31;
    const int half = lane >> 4;
    const int ln   = lane & 15;

    // Prime the pipeline: B tile for k0=0 into buffer 0 while we stage stats.
    if (wave == 0) tdm_issue(0, 0);

    for (int i = t; i < IN_F; i += 256) {
        s_mu [i] = mu[i];
        s_thr[i] = thresholds[(size_t)r * IN_F + i] * stdv[i];
    }
    if (t < ROWS) s_cnt[t] = 0;

    const int arow = t >> 3;          // A staging: 1 row, 4 consecutive k
    const int akk  = (t & 7) * 4;
    const float* xrow = x + (size_t)(nbase + arow) * IN_F + akk;

    v8f acc00 = {}, acc01 = {}, acc10 = {}, acc11 = {};
    int cnt = 0;

    __syncthreads();

    for (int k0 = 0; k0 < IN_F; k0 += KTILE) {
        const int cur = (k0 >> 5) & 1;

        // ---- stage A: centered, threshold-masked activations -> bf16 ----
        {
            const float4 xv = *reinterpret_cast<const float4*>(xrow + k0);
            const float f[4] = {xv.x, xv.y, xv.z, xv.w};
            v4bf ah;
#pragma unroll
            for (int j = 0; j < 4; ++j) {
                const int   gi = k0 + akk + j;
                const float xm = f[j] - s_mu[gi];
                const bool  m  = __builtin_fabsf(xm) > s_thr[gi];
                cnt += m ? 1 : 0;
                ah[j] = m ? (bf16_t)xm : (bf16_t)0.0f;
            }
            *reinterpret_cast<v4bf*>(&s_A[arow * ASTRIDE + akk]) = ah;
            if (k0 + KTILE < IN_F)
                __builtin_prefetch(xrow + k0 + KTILE, 0, 0);
        }

        // ---- wave 0: launch TDM for next tile, wait for current tile ----
        if (wave == 0) {
            if (k0 + KTILE < IN_F) {
                tdm_issue(k0 + KTILE, cur ^ 1);
                __builtin_amdgcn_s_wait_tensorcnt(1);   // oldest (cur) done
            } else {
                __builtin_amdgcn_s_wait_tensorcnt(0);
            }
        }
        __syncthreads();

        // ---- A fragments (16 contiguous K per K-half) ----
        v16bf afrag[2];
#pragma unroll
        for (int rt = 0; rt < 2; ++rt) {
            const bf16_t* ab = &s_A[(rt * 16 + ln) * ASTRIDE + half * 16];
            const v8bf c0 = *reinterpret_cast<const v8bf*>(ab);      // K = half*16..
            const v8bf c1 = *reinterpret_cast<const v8bf*>(ab + 8);  // K = half*16+8..
            v16bf tmp;
#pragma unroll
            for (int i = 0; i < 8; ++i) { tmp[i] = c0[i]; tmp[8 + i] = c1[i]; }
            afrag[rt] = tmp;
        }

        // ---- B fragments: column per lane, 16 contiguous K per K-half ----
        v16bf bfrag[2];
#pragma unroll
        for (int ct = 0; ct < 2; ++ct) {
            const int scol = (wave * 2 + ct) * 16 + ln;
            v16bf tmp;
            if constexpr (PRE) {
                const bf16_t* bb = (const bf16_t*)&s_Braw[cur * BUFBYTES] +
                                   scol * BSTR + half * 16;
                const v8bf q0 = *reinterpret_cast<const v8bf*>(bb);
                const v8bf q1 = *reinterpret_cast<const v8bf*>(bb + 8);
#pragma unroll
                for (int i = 0; i < 8; ++i) { tmp[i] = q0[i]; tmp[8 + i] = q1[i]; }
            } else {
                const float* fB = (const float*)&s_Braw[cur * BUFBYTES];
#pragma unroll
                for (int e = 0; e < 16; ++e)
                    tmp[e] = (bf16_t)fB[(half * 16 + e) * STRIPE_W + scol];
            }
            bfrag[ct] = tmp;
        }

        // ---- 4x v_wmma_f32_16x16x32_bf16 with fragment reuse ----
        acc00 = __builtin_amdgcn_wmma_f32_16x16x32_bf16(false, afrag[0], false, bfrag[0],
                                                        (short)0, acc00, false, false);
        acc01 = __builtin_amdgcn_wmma_f32_16x16x32_bf16(false, afrag[0], false, bfrag[1],
                                                        (short)0, acc01, false, false);
        acc10 = __builtin_amdgcn_wmma_f32_16x16x32_bf16(false, afrag[1], false, bfrag[0],
                                                        (short)0, acc10, false, false);
        acc11 = __builtin_amdgcn_wmma_f32_16x16x32_bf16(false, afrag[1], false, bfrag[1],
                                                        (short)0, acc11, false, false);
        __syncthreads();
    }

    // ---- mask popcount reduction (deterministic integer atomics) ----
    atomicAdd(&s_cnt[arow], cnt);

    // ---- epilogue: add (mu@W + bias), store C tiles per ISA layout ----
#pragma unroll
    for (int ct = 0; ct < 2; ++ct) {
        const int   col = r * STRIPE_W + (wave * 2 + ct) * 16 + ln;
        const float p   = pmb[col];
        const v8f   a0  = (ct == 0) ? acc00 : acc01;
        const v8f   a1  = (ct == 0) ? acc10 : acc11;
#pragma unroll
        for (int v = 0; v < 8; ++v) {
            const int row0 = nbase + half * 8 + v;       // M = v (+8 for hi lanes)
            y[(size_t)row0 * OUT_F + col]        = a0[v] + p;
            y[(size_t)(row0 + 16) * OUT_F + col] = a1[v] + p;
        }
    }

    __syncthreads();
    if (t < ROWS) atomicAdd(&counts[nbase + t], s_cnt[t]);
}

// --------------------------- finalize flop outputs -------------------------
__global__ __launch_bounds__(256) void cwic_finalize(
    const int* __restrict__ counts, float* __restrict__ dense,
    float* __restrict__ sparse)
{
    const int n = blockIdx.x * 256 + threadIdx.x;
    if (n < NTOK) {
        dense[n]  = 16777216.0f;                  // I*O
        sparse[n] = 256.0f * (float)counts[n];    // (I*O)/(ns*I) * sum(mask)
    }
}

// ---------------------------------------------------------------------------
extern "C" void kernel_launch(void* const* d_in, const int* in_sizes, int n_in,
                              void* d_out, int out_size, void* d_ws, size_t ws_size,
                              hipStream_t stream) {
    (void)in_sizes; (void)n_in; (void)out_size;
    const float* x      = (const float*)d_in[0];
    const float* weight = (const float*)d_in[1];
    const float* bias   = (const float*)d_in[2];
    const float* mu     = (const float*)d_in[3];
    const float* stdv   = (const float*)d_in[4];
    const float* thr    = (const float*)d_in[5];

    float* y      = (float*)d_out;
    float* dense  = y + (size_t)NTOK * OUT_F;
    float* sparse = dense + NTOK;

    int*    counts = (int*)d_ws;                                   // 8 KB
    float*  pmb    = (float*)((char*)d_ws + 8192);                 // 32 KB
    bf16_t* wt     = (bf16_t*)((char*)d_ws + 40960);               // 32 MB (optional)
    const size_t need_pre = 40960 + (size_t)IN_F * OUT_F * sizeof(bf16_t);
    const bool   pre      = ws_size >= need_pre;

    hipLaunchKernelGGL(cwic_prep, dim3(OUT_F / 256), dim3(256), 0, stream,
                       weight, bias, mu, pmb, counts);
    if (pre) {
        hipLaunchKernelGGL(cwic_wt_bf16, dim3(OUT_F / 64, IN_F / 32), dim3(256),
                           0, stream, weight, wt);
        hipLaunchKernelGGL(cwic_stripe_gemm<true>, dim3(NSTRIPES, NTOK / ROWS),
                           dim3(256), 0, stream,
                           x, weight, wt, mu, stdv, thr, pmb, counts, y);
    } else {
        hipLaunchKernelGGL(cwic_stripe_gemm<false>, dim3(NSTRIPES, NTOK / ROWS),
                           dim3(256), 0, stream,
                           x, weight, wt, mu, stdv, thr, pmb, counts, y);
    }
    hipLaunchKernelGGL(cwic_finalize, dim3(NTOK / 256), dim3(256), 0, stream,
                       counts, dense, sparse);
}
